// AugmentedLstm_10565619549065
// MI455X (gfx1250) — matrix operand order repaired and checked by reading, not compiled, not run on hardware
//
#include <hip/hip_runtime.h>

// ---------------------------------------------------------------------------
// Augmented LSTM for MI455X (gfx1250, wave32, WMMA, async global->LDS).
//
// Per step t (one kernel, 48 workgroups): each WG owns hidden slice
// j in [j0, j0+16) and computes the six pre-activation column strips
// {g*H + j0 .. g*H + j0 + 15}, g = 0..5, as a 64x96 GEMM tile:
//     z = x_t @ w_in^T + b_in   (+ h @ w_state^T + b_state for g < 5)
// using bf16 hi/lo split (3 WMMAs per K-slice, fp32 accumulate) for
// ~fp32 accuracy at bf16 matrix rates. Weights are pre-split to bf16
// hi/lo once per call, so per-step B staging is a pure async copy
// (global_load_async_to_lds_b128, ASYNCcnt). Gates are evaluated in the
// same kernel from LDS — no z round-trip through HBM, 1 launch/step.
// ---------------------------------------------------------------------------

typedef __attribute__((ext_vector_type(16))) __bf16 bf16x16;
typedef __attribute__((ext_vector_type(8)))  float  f32x8;

namespace {

constexpr int kB = 64, kT = 512, kE = 768, kH = 768;
constexpr int KC  = 64;        // K chunk per stage (2 bf16 WMMA K-slices)
constexpr int LDA = KC + 8;    // 72 bf16 = 144 B rows: 16B-aligned, bank-skewed
constexpr int LDB = KC + 8;
constexpr int JW  = 16;        // hidden columns per workgroup
constexpr int NCOL = 6 * JW;   // 96 staged B rows (6 gate strips)

union FragU { bf16x16 bf; uint4 q[2]; };

// A fragment (16x32 bf16, ISA 7.12.2): lanes 0-15: M=lane, K {0-7,16-23};
// lanes 16-31: M=lane-16, K {8-15,24-31}. Two aligned b128 LDS loads.
__device__ inline bf16x16 load_a_frag(const __bf16* As, int mBase, int kh,
                                      int lane) {
  const uint4* p =
      reinterpret_cast<const uint4*>(As + (mBase + (lane & 15)) * LDA);
  const int g = lane >> 4;
  FragU r;
  r.q[0] = p[kh * 4 + g];          // K = kh*32 + {0..7 | 8..15}
  r.q[1] = p[kh * 4 + 2 + g];      // K = kh*32 + 16 + {0..7 | 8..15}
  return r.bf;
}

// B fragment (32x16 bf16): lanes 0-15 -> K {0-7,16-23}, lanes 16-31 -> K {8-15,24-31},
// N = lane&15. LDS holds B as [n][k], so K is contiguous: two b128 loads.
__device__ inline bf16x16 load_b_frag(const __bf16* Bs, int nBase, int kh,
                                      int lane) {
  const uint4* p =
      reinterpret_cast<const uint4*>(Bs + (nBase + (lane & 15)) * LDB);
  const int g = lane >> 4;
  FragU r;
  r.q[0] = p[kh * 4 + g];
  r.q[1] = p[kh * 4 + 2 + g];
  return r.bf;
}

__device__ inline f32x8 wmma_bf16(bf16x16 a, bf16x16 b, f32x8 c) {
  return __builtin_amdgcn_wmma_f32_16x16x32_bf16(false, a, false, b, (short)0,
                                                 c, false, false);
}

__device__ inline void cvt_split2(float2 v, unsigned& hiP, unsigned& loP) {
  union { __bf16 b[2]; unsigned u; } h, l;
  h.b[0] = (__bf16)v.x; l.b[0] = (__bf16)(v.x - (float)h.b[0]);
  h.b[1] = (__bf16)v.y; l.b[1] = (__bf16)(v.y - (float)h.b[1]);
  hiP = h.u; loP = l.u;
}

// CDNA5 async global->LDS copy (ASYNCcnt-tracked), 16 B per lane.
__device__ inline void async_copy_b128(unsigned lds_off,
                                       unsigned long long gaddr) {
  asm volatile("global_load_async_to_lds_b128 %0, %1, off" ::"v"(lds_off),
               "v"(gaddr)
               : "memory");
}
__device__ inline void wait_async0() {
  asm volatile("s_wait_asynccnt 0x0" ::: "memory");
}

__device__ inline float sigmoidf(float x) { return 1.0f / (1.0f + __expf(-x)); }

__global__ __launch_bounds__(256) void step_fused(
    const float* __restrict__ x, const __bf16* __restrict__ win_hi,
    const __bf16* __restrict__ win_lo, const __bf16* __restrict__ wst_hi,
    const __bf16* __restrict__ wst_lo, const float* __restrict__ b_in,
    const float* __restrict__ b_state, const long long* __restrict__ lengths,
    const float* __restrict__ h_cur, const float* __restrict__ c_cur,
    float* __restrict__ h_nxt, float* __restrict__ c_nxt,
    float* __restrict__ ys, int t) {
  __shared__ __attribute__((aligned(16))) __bf16 As_hi[64 * LDA];
  __shared__ __attribute__((aligned(16))) __bf16 As_lo[64 * LDA];
  __shared__ __attribute__((aligned(16))) __bf16 Bs_hi[NCOL * LDB];
  __shared__ __attribute__((aligned(16))) __bf16 Bs_lo[NCOL * LDB];
  __shared__ float zloc[64 * NCOL];

  const int tid = threadIdx.x, wave = tid >> 5, lane = tid & 31;
  const int j0 = blockIdx.x * JW;

  // Wave w owns tiles tau = 3w..3w+2; tau -> (strip = tau>>2, mt = tau&3).
  f32x8 acc[3];
#pragma unroll
  for (int ti = 0; ti < 3; ++ti)
#pragma unroll
    for (int r = 0; r < 8; ++r) acc[ti][r] = 0.0f;

  for (int phase = 0; phase < 2; ++phase) {
    const float*  Asrc = phase ? h_cur : (x + (size_t)t * kE);
    const size_t  aStr = phase ? (size_t)kH : (size_t)kT * kE;
    const __bf16* Whi  = phase ? wst_hi : win_hi;
    const __bf16* Wlo  = phase ? wst_lo : win_lo;
    const int     rows = phase ? 5 * JW : NCOL;  // strips with w_state: 0..4

    for (int kc = 0; kc < kE; kc += KC) {
      // Stage A (64 x 64 f32) -> bf16 hi/lo, pair-packed dword stores.
      for (int i = tid; i < 64 * (KC / 2); i += 256) {
        const int r = i >> 5, kp = (i & 31) * 2;
        const float2 v = *reinterpret_cast<const float2*>(
            Asrc + (size_t)r * aStr + (kc + kp));
        unsigned hp, lp;
        cvt_split2(v, hp, lp);
        reinterpret_cast<unsigned*>(As_hi)[r * (LDA / 2) + (kp >> 1)] = hp;
        reinterpret_cast<unsigned*>(As_lo)[r * (LDA / 2) + (kp >> 1)] = lp;
      }
      // Stage B: async 16B copies of pre-split bf16 weights.
      for (int i = tid; i < rows * (KC / 8); i += 256) {
        const int r = i >> 3, kq = (i & 7) * 8;
        const int grow = (r >> 4) * kH + j0 + (r & 15);
        const size_t go = (size_t)grow * kE + (kc + kq);
        async_copy_b128((unsigned)(size_t)(void*)&Bs_hi[r * LDB + kq],
                        (unsigned long long)(size_t)(Whi + go));
        async_copy_b128((unsigned)(size_t)(void*)&Bs_lo[r * LDB + kq],
                        (unsigned long long)(size_t)(Wlo + go));
      }
      wait_async0();
      __syncthreads();

#pragma unroll
      for (int ti = 0; ti < 3; ++ti) {
        const int tau = wave * 3 + ti;
        const int strip = tau >> 2, mt = tau & 3;
        if (phase == 1 && strip >= 5) continue;  // no w_state on strip 5
#pragma unroll
        for (int kh = 0; kh < 2; ++kh) {
          const bf16x16 bhi = load_b_frag(Bs_hi, strip * 16, kh, lane);
          const bf16x16 blo = load_b_frag(Bs_lo, strip * 16, kh, lane);
          const bf16x16 ahi = load_a_frag(As_hi, mt * 16, kh, lane);
          const bf16x16 alo = load_a_frag(As_lo, mt * 16, kh, lane);
          acc[ti] = wmma_bf16(ahi, bhi, acc[ti]);  // hi*hi
          acc[ti] = wmma_bf16(alo, bhi, acc[ti]);  // lo*hi
          acc[ti] = wmma_bf16(ahi, blo, acc[ti]);  // hi*lo
        }
      }
      __syncthreads();
    }
  }

  // Spill accumulators (+bias) to LDS. C/D layout: lanes 0-15 M=r,
  // lanes 16-31 M=8+r, N=lane&15.
  const int rowOff = (lane >> 4) * 8;
#pragma unroll
  for (int ti = 0; ti < 3; ++ti) {
    const int tau = wave * 3 + ti;
    const int strip = tau >> 2, mt = tau & 3;
    const int n = strip * kH + j0 + (lane & 15);
    const float bias = b_in[n] + ((strip < 5) ? b_state[n] : 0.0f);
#pragma unroll
    for (int r = 0; r < 8; ++r)
      zloc[(mt * 16 + rowOff + r) * NCOL + strip * 16 + (lane & 15)] =
          acc[ti][r] + bias;
  }
  __syncthreads();

  // Gates: all 6 strips for j in [j0, j0+16) are local (4 items / thread).
  for (int i = tid; i < 64 * JW; i += 256) {
    const int b = i >> 4, jl = i & 15;
    const float* z = zloc + b * NCOL;
    const float ig = sigmoidf(z[jl]);
    const float fg = sigmoidf(z[16 + jl]);
    const float mi = tanhf(z[32 + jl]);
    const float og = sigmoidf(z[48 + jl]);
    const float hg = sigmoidf(z[64 + jl]);
    const float pex = z[80 + jl];
    const int gidx = b * kH + j0 + jl;
    const float c_old = c_cur[gidx], h_old = h_cur[gidx];
    const float c_new = ig * mi + fg * c_old;
    float out = og * tanhf(c_new);
    out = hg * out + (1.0f - hg) * pex;
    const float m = (t < (int)lengths[b]) ? 1.0f : 0.0f;
    const float y = m * out;
    ys[((size_t)b * kT + t) * kH + (j0 + jl)] = y;
    h_nxt[gidx] = y + (1.0f - m) * h_old;
    c_nxt[gidx] = m * c_new + (1.0f - m) * c_old;
  }
}

// One-time per call: split fp32 weights into bf16 hi/lo (row-major [n][k]).
__global__ __launch_bounds__(256) void prep_weights(
    const float* __restrict__ w_in, const float* __restrict__ w_state,
    __bf16* __restrict__ win_hi, __bf16* __restrict__ win_lo,
    __bf16* __restrict__ wst_hi, __bf16* __restrict__ wst_lo) {
  const size_t nin = (size_t)(6 * kH) * kE;
  const size_t nst = (size_t)(5 * kH) * kH;
  const size_t i = (size_t)blockIdx.x * 256 + threadIdx.x;
  if (i < nin) {
    const float v = w_in[i];
    const __bf16 hi = (__bf16)v;
    win_hi[i] = hi;
    win_lo[i] = (__bf16)(v - (float)hi);
  }
  if (i < nst) {
    const float v = w_state[i];
    const __bf16 hi = (__bf16)v;
    wst_hi[i] = hi;
    wst_lo[i] = (__bf16)(v - (float)hi);
  }
}

__global__ __launch_bounds__(256) void init_state(const float* __restrict__ h0,
                                                  const float* __restrict__ c0,
                                                  float* __restrict__ hb,
                                                  float* __restrict__ cb) {
  const int i = blockIdx.x * 256 + threadIdx.x;
  if (i < kB * kH) { hb[i] = h0[i]; cb[i] = c0[i]; }
}

__global__ __launch_bounds__(256) void fin_state(const float* __restrict__ hb,
                                                 const float* __restrict__ cb,
                                                 float* __restrict__ hT,
                                                 float* __restrict__ cT) {
  const int i = blockIdx.x * 256 + threadIdx.x;
  if (i < kB * kH) { hT[i] = hb[i]; cT[i] = cb[i]; }
}

}  // namespace

extern "C" void kernel_launch(void* const* d_in, const int* in_sizes, int n_in,
                              void* d_out, int out_size, void* d_ws,
                              size_t ws_size, hipStream_t stream) {
  (void)in_sizes; (void)n_in; (void)out_size; (void)ws_size;

  const float*     x       = (const float*)d_in[0];
  const long long* lengths = (const long long*)d_in[1];  // int64
  const float*     w_in    = (const float*)d_in[2];
  const float*     b_in    = (const float*)d_in[3];
  const float*     w_state = (const float*)d_in[4];
  const float*     b_state = (const float*)d_in[5];
  const float*     h0      = (const float*)d_in[6];
  const float*     c0      = (const float*)d_in[7];

  float* out = (float*)d_out;
  float* ys  = out;                           // (B, T, H)
  float* hT  = out + (size_t)kB * kT * kH;    // (1, B, H)
  float* cT  = hT + (size_t)kB * kH;          // (1, B, H)

  // Workspace: h/c double buffers + pre-split bf16 weights (~26.8 MB).
  const size_t BH = (size_t)kB * kH;
  float*  hb     = (float*)d_ws;              // 2 * B*H
  float*  cb     = hb + 2 * BH;               // 2 * B*H
  __bf16* win_hi = (__bf16*)(cb + 2 * BH);
  __bf16* win_lo = win_hi + (size_t)(6 * kH) * kE;
  __bf16* wst_hi = win_lo + (size_t)(6 * kH) * kE;
  __bf16* wst_lo = wst_hi + (size_t)(5 * kH) * kH;

  const size_t nin = (size_t)(6 * kH) * kE;
  prep_weights<<<(int)((nin + 255) / 256), 256, 0, stream>>>(
      w_in, w_state, win_hi, win_lo, wst_hi, wst_lo);

  const int bh_blocks = (int)(BH / 256);  // 192
  init_state<<<bh_blocks, 256, 0, stream>>>(h0, c0, hb, cb);

  for (int t = 0; t < kT; ++t) {
    const int    cur  = t & 1, nxt = cur ^ 1;
    const float* hcur = hb + (size_t)cur * BH;
    const float* ccur = cb + (size_t)cur * BH;
    float*       hnxt = hb + (size_t)nxt * BH;
    float*       cnxt = cb + (size_t)nxt * BH;
    step_fused<<<kH / JW, 256, 0, stream>>>(x, win_hi, win_lo, wst_hi, wst_lo,
                                            b_in, b_state, lengths, hcur, ccur,
                                            hnxt, cnxt, ys, t);
  }

  // T is even: final state lives in buffer 0.
  fin_state<<<bh_blocks, 256, 0, stream>>>(hb, cb, hT, cT);
}
